// TactileVQ_27401891348683
// MI455X (gfx1250) — matrix operand order repaired and checked by reading, not compiled, not run on hardware
//
#include <hip/hip_runtime.h>
#include <string.h>

// ---------------------------------------------------------------------------
// VQ codebook assignment on MI455X (gfx1250):
//   d2[n,k] = ||z_n||^2 - 2 z_n.e_k + ||e_k||^2 ; argmin_k ; gather ; MSE
// GEMM part done with V_WMMA_F32_16X16X32_F16 (f16 inputs, f32 accum).
// Codebook staged in LDS as f16 (256KB + 4KB e_sq < 320KB/WGP).
// ---------------------------------------------------------------------------

typedef __attribute__((ext_vector_type(16))) _Float16 v16h;
typedef __attribute__((ext_vector_type(4)))  _Float16 v4h;
typedef __attribute__((ext_vector_type(8)))  float    v8f;

#define N_PTS 131072
#define K_CB  1024
#define D_DIM 128

#define LDS_ESQ_BYTES 4096
#define LDS_EMB_BYTES (K_CB * D_DIM * 2)
#define LDS_BYTES     (LDS_ESQ_BYTES + LDS_EMB_BYTES)

extern "C" __global__ __launch_bounds__(256)
void vq_assign_wmma(const float* __restrict__ z_e,
                    const float* __restrict__ emb,
                    float* __restrict__ zq_out,
                    float* __restrict__ loss_out,
                    float* __restrict__ idx_out)
{
    extern __shared__ char smem[];
    float*    esq = (float*)smem;                       // [1024] f32
    _Float16* eh  = (_Float16*)(smem + LDS_ESQ_BYTES);  // [1024][128] f16 row-major

    const int tid = threadIdx.x;

    // ---- Stage codebook to LDS (f32 -> f16) and compute ||e_k||^2 ----------
    // 256 threads, 4 codewords each.
    for (int c = 0; c < 4; ++c) {
        const int cw = tid * 4 + c;
        const float4* src = (const float4*)(emb + (size_t)cw * D_DIM);
        float s = 0.0f;
        _Float16* dst = eh + (size_t)cw * D_DIM;
#pragma unroll
        for (int i = 0; i < 32; ++i) {
            float4 v = src[i];
            s += v.x * v.x + v.y * v.y + v.z * v.z + v.w * v.w;
            v4h h = { (_Float16)v.x, (_Float16)v.y, (_Float16)v.z, (_Float16)v.w };
            *(v4h*)(dst + i * 4) = h;
        }
        esq[cw] = s;
    }
    __syncthreads();

    // ---- Per-wave 16-row tile -----------------------------------------------
    const int lane = tid & 31;
    const int wave = tid >> 5;
    const int half = lane >> 4;   // 0: lanes 0-15, 1: lanes 16-31
    const int m    = lane & 15;

    const int row_base = (blockIdx.x * 8 + wave) * 16;

    // A fragment: 16x128 f16. Contiguous k order (matching B) per 32-wide
    // k-step j: lane (M = m) holds K = j*32 + half*16 + [0..7] in halfs 0..7
    // and K = j*32 + half*16 + 8 + [0..7] in halfs 8..15.
    const float* zrow = z_e + (size_t)(row_base + m) * D_DIM;
    v16h a[4];
#pragma unroll
    for (int j = 0; j < 4; ++j) {
        const float* c1 = zrow + j * 32 + half * 16;
        const float* c2 = zrow + j * 32 + half * 16 + 8;
#pragma unroll
        for (int t = 0; t < 8; ++t) {
            a[j][t]     = (_Float16)c1[t];
            a[j][8 + t] = (_Float16)c2[t];
        }
    }

    // Min / argmin trackers: C layout element (M = half*8 + r, N = m).
    float minv[8];
    int   mini[8];
#pragma unroll
    for (int r = 0; r < 8; ++r) { minv[r] = 3.402823466e38f; mini[r] = 0; }

    // ---- Sweep all 64 codeword tiles ---------------------------------------
#pragma unroll 1
    for (int t = 0; t < 64; ++t) {
        const int cw = t * 16 + m;   // this lane's B column / codeword
        v8f acc = {};
        // B 32x16 layout: lanes 0-15 hold K=0..15, lanes 16-31 hold K=16..31,
        // contiguous halfs -> one 32B LDS read per k-step.
#pragma unroll
        for (int j = 0; j < 4; ++j) {
            v16h b = *(const v16h*)(eh + (size_t)cw * D_DIM + j * 32 + half * 16);
            acc = __builtin_amdgcn_wmma_f32_16x16x32_f16(
                false, a[j], false, b, (short)0, acc, false, false);
        }
        const float es = esq[cw];
#pragma unroll
        for (int r = 0; r < 8; ++r) {
            const float d2 = es - 2.0f * acc[r];   // ||z||^2 dropped (row-const)
            if (d2 < minv[r]) { minv[r] = d2; mini[r] = cw; }
        }
    }

    // ---- Argmin butterfly across the 16 lanes holding each row -------------
#pragma unroll
    for (int r = 0; r < 8; ++r) {
#pragma unroll
        for (int s = 1; s < 16; s <<= 1) {
            float ov = __shfl_xor(minv[r], s, 32);
            int   oi = __shfl_xor(mini[r], s, 32);
            if (ov < minv[r] || (ov == minv[r] && oi < mini[r])) {
                minv[r] = ov; mini[r] = oi;
            }
        }
    }

    // Write indices (as float, per output dtype). Lanes 0-15 hold rows 0-7,
    // lanes 16-31 hold rows 8-15; one writer per group.
    if (m == 0) {
#pragma unroll
        for (int r = 0; r < 8; ++r)
            idx_out[row_base + half * 8 + r] = (float)mini[r];
    }

    // ---- Gather z_q = emb[idx] (f32, L2-hot) + commitment loss -------------
    float lsum = 0.0f;
#pragma unroll
    for (int rr = 0; rr < 16; ++rr) {
        const int src_lane = (rr >> 3) * 16;
        const int idx = __shfl(mini[rr & 7], src_lane, 32);
        const float4* esrc = (const float4*)(emb + (size_t)idx * D_DIM) + lane;
        const float4* zsrc = (const float4*)(z_e + (size_t)(row_base + rr) * D_DIM) + lane;
        float4 e4 = *esrc;
        float4 z4 = *zsrc;
        ((float4*)(zq_out + (size_t)(row_base + rr) * D_DIM))[lane] = e4;
        const float dx = e4.x - z4.x, dy = e4.y - z4.y;
        const float dz = e4.z - z4.z, dw = e4.w - z4.w;
        lsum += dx * dx + dy * dy + dz * dz + dw * dw;
    }
#pragma unroll
    for (int s = 1; s < 32; s <<= 1) lsum += __shfl_xor(lsum, s, 32);
    if (lane == 0)
        atomicAdd(loss_out, lsum * (1.0f / ((float)N_PTS * (float)D_DIM)));
}

// ---------------------------------------------------------------------------
extern "C" void kernel_launch(void* const* d_in, const int* in_sizes, int n_in,
                              void* d_out, int out_size, void* d_ws, size_t ws_size,
                              hipStream_t stream)
{
    (void)in_sizes; (void)n_in; (void)out_size; (void)d_ws; (void)ws_size;

    const float* z_e = (const float*)d_in[0];
    const float* emb = (const float*)d_in[1];

    float* zq   = (float*)d_out;                       // [N, D] f32
    float* loss = zq + (size_t)N_PTS * D_DIM;          // [1]
    float* idxf = loss + 1;                            // [N] (indices as f32)

    // Loss slot accumulated via atomics -> zero it (capture-safe memset).
    hipMemsetAsync(loss, 0, sizeof(float), stream);

    // Allow >64KB dynamic LDS (best effort; harmless if already permitted).
    (void)hipFuncSetAttribute((const void*)vq_assign_wmma,
                              hipFuncAttributeMaxDynamicSharedMemorySize,
                              (int)LDS_BYTES);

    const int blocks = N_PTS / 128;   // 8 waves/block x 16 rows/wave = 128 rows
    vq_assign_wmma<<<blocks, 256, LDS_BYTES, stream>>>(z_e, emb, zq, loss, idxf);
}
